// GAT_90615220011132
// MI455X (gfx1250) — matrix operand order repaired and checked by reading, not compiled, not run on hardware
//
#include <hip/hip_runtime.h>
#include <hip/hip_bf16.h>

// ---------------- problem constants (from reference) ----------------
#define GN 50000          // nodes (divisible by 16)
#define GE 600000         // directed edges before self loops
#define GEP (GE + GN)     // edges + self loops
#define GB 128            // graphs
#define GD 128            // channels
#define GH 4              // heads
#define GDH 32            // dims per head
#define NEG_SLOPE 0.2f

typedef __attribute__((ext_vector_type(16))) _Float16 v16h;
typedef __attribute__((ext_vector_type(8)))  float    v8f;

// ---- monotonic float<->uint encoding for atomic max over signed floats ----
__device__ __forceinline__ unsigned enc_f(float f) {
    unsigned u = __float_as_uint(f);
    return (u & 0x80000000u) ? ~u : (u | 0x80000000u);
}
__device__ __forceinline__ float dec_f(unsigned u) {
    return (u & 0x80000000u) ? __uint_as_float(u ^ 0x80000000u)
                             : __uint_as_float(~u);
}

// ---------------- h0 = emb[x]  (float4 per thread) ----------------
__global__ void k_init_h(const int* __restrict__ x, const float* __restrict__ emb,
                         float* __restrict__ h) {
    int t = blockIdx.x * 256 + threadIdx.x;
    if (t >= GN * 32) return;
    int n = t >> 5, q = t & 31;
    const float4* e4 = (const float4*)(emb + (size_t)x[n] * GD);
    ((float4*)h)[t] = e4[q];
}

// ---------------- Wt16[c][k] = (f16) W[k][c]  (transpose + downconvert) ----
__global__ void k_cvt_wt(const float* __restrict__ W, _Float16* __restrict__ wt) {
    int t = blockIdx.x * 256 + threadIdx.x;
    if (t >= GD * GD) return;
    int k = t >> 7, c = t & 127;
    wt[c * GD + k] = (_Float16)W[t];
}

// ---------------- z = h @ W  via V_WMMA_F32_16X16X32_F16 ----------------
// block = 256 threads = 8 wave32; block handles a 16-row strip of h;
// wave w computes the 16x16 output tile at columns [16w, 16w+16).
// B comes from the pre-transposed f16 weight: per-lane fragments are 32
// contiguous bytes -> two global_load_b128, no converts in the hot loop.
__global__ void k_gemm_z(const float* __restrict__ h, const _Float16* __restrict__ wt,
                         float* __restrict__ z) {
    __shared__ _Float16 As[16 * 128];       // f16 staged A strip, 4 KB
    const int lane = threadIdx.x & 31;
    const int wave = threadIdx.x >> 5;      // n-tile id 0..7
    const int m0   = blockIdx.x * 16;

    // stage + convert 16x128 fp32 -> f16 into LDS
    for (int i = threadIdx.x; i < 16 * 128; i += 256)
        As[i] = (_Float16)h[(size_t)m0 * 128 + i];
    __syncthreads();

    const int hi   = lane >> 4;             // half-wave select
    const int mrow = lane & 15;             // A row for this lane
    const int ncol = lane & 15;             // B/D column for this lane
    const int n0   = wave * 16;

    v8f acc = {};
    #pragma unroll
    for (int kt = 0; kt < 4; ++kt) {        // K = 128 = 4 x 32
        const int kb = kt * 32;
        v16h a;
        #pragma unroll
        for (int j = 0; j < 16; ++j) {
            // A 16x32 f16 layout: element j of the lane holds K = j + 16*hi
            // (contiguous order, matching B)
            int ka = j + 16 * hi;
            a[j] = As[mrow * 128 + kb + ka];
        }
        // B 32x16 f16 layout: lane = column n0+ncol; K = j + 16*hi,
        // contiguous in the transposed f16 weight (32B aligned).
        v16h b = *(const v16h*)(wt + (size_t)(n0 + ncol) * GD + kb + 16 * hi);
        acc = __builtin_amdgcn_wmma_f32_16x16x32_f16(false, a, false, b,
                                                     (short)0, acc, false, false);
    }
    // C/D layout: VGPR i -> row m0 + i + 8*hi, column n0 + ncol
    #pragma unroll
    for (int i = 0; i < 8; ++i)
        z[(size_t)(m0 + i + 8 * hi) * 128 + n0 + ncol] = acc[i];
}

// ---------------- per-node attention logits ----------------
__global__ void k_attn_logits(const float* __restrict__ z,
                              const float* __restrict__ asrc,
                              const float* __restrict__ adst,
                              float* __restrict__ a_s, float* __restrict__ a_d) {
    int t = blockIdx.x * 256 + threadIdx.x;
    if (t >= GN * GH) return;
    int n = t >> 2, hd = t & 3;
    const float* zr = z + (size_t)n * GD + hd * GDH;
    float ss = 0.f, sd = 0.f;
    #pragma unroll
    for (int k = 0; k < GDH; ++k) {
        float v = zr[k];
        ss += v * asrc[hd * GDH + k];
        sd += v * adst[hd * GDH + k];
    }
    a_s[t] = ss;
    a_d[t] = sd;
}

__device__ __forceinline__ void edge_sd(const int* ei, int e, int& s, int& d) {
    if (e < GE) { s = ei[e]; d = ei[GE + e]; }
    else        { s = d = e - GE; }           // self loop
}

// ---------------- segment max of leaky_relu(alpha) over dst ----------------
__global__ void k_edge_max(const int* __restrict__ ei,
                           const float* __restrict__ a_s, const float* __restrict__ a_d,
                           unsigned* __restrict__ amax) {
    int t = blockIdx.x * 256 + threadIdx.x;
    if (t >= GEP * GH) return;
    int e = t >> 2, hd = t & 3, s, d;
    edge_sd(ei, e, s, d);
    float a = a_s[s * GH + hd] + a_d[d * GH + hd];
    a = (a > 0.f) ? a : NEG_SLOPE * a;
    atomicMax(&amax[d * GH + hd], enc_f(a));
}

// ---------------- exp + segment sum ----------------
__global__ void k_edge_exp(const int* __restrict__ ei,
                           const float* __restrict__ a_s, const float* __restrict__ a_d,
                           const unsigned* __restrict__ amax,
                           float* __restrict__ den, float* __restrict__ exv) {
    int t = blockIdx.x * 256 + threadIdx.x;
    if (t >= GEP * GH) return;
    int e = t >> 2, hd = t & 3, s, d;
    edge_sd(ei, e, s, d);
    float a = a_s[s * GH + hd] + a_d[d * GH + hd];
    a = (a > 0.f) ? a : NEG_SLOPE * a;
    float ex = __expf(a - dec_f(amax[d * GH + hd]));
    atomicAdd(&den[d * GH + hd], ex);
    exv[t] = ex;
}

// ---------------- coef = ex/den + 1 (f_additive) ----------------
__global__ void k_edge_coef(const int* __restrict__ ei,
                            const float* __restrict__ den, float* __restrict__ exv) {
    int t = blockIdx.x * 256 + threadIdx.x;
    if (t >= GEP * GH) return;
    int e = t >> 2, hd = t & 3, s, d;
    edge_sd(ei, e, s, d);
    exv[t] = exv[t] / den[d * GH + hd] + 1.0f;
}

// -------- out[dst] += z[src] * coef  (L2-resident scatter, float4/thread) ---
__global__ void k_edge_scatter(const int* __restrict__ ei,
                               const float* __restrict__ z,
                               const float* __restrict__ coef,
                               float* __restrict__ out) {
    int t = blockIdx.x * 256 + threadIdx.x;
    if (t >= GEP * 32) return;
    int e = t >> 5, q = t & 31, d = q * 4, s, dd;
    edge_sd(ei, e, s, dd);
    float c = coef[e * GH + (d >> 5)];             // one head per float4 chunk
    float4 zv = *(const float4*)(z + (size_t)s * GD + d);
    float* o = out + (size_t)dd * GD + d;
    atomicAdd(o + 0, zv.x * c);
    atomicAdd(o + 1, zv.y * c);
    atomicAdd(o + 2, zv.z * c);
    atomicAdd(o + 3, zv.w * c);
}

// ---------------- h' = elu(out) + h   (in place on out, float4) ------------
__global__ void k_elu_res(float* __restrict__ out, const float* __restrict__ h) {
    int t = blockIdx.x * 256 + threadIdx.x;
    if (t >= GN * 32) return;
    float4 v = ((const float4*)out)[t];
    float4 r = ((const float4*)h)[t];
    v.x = ((v.x > 0.f) ? v.x : (__expf(v.x) - 1.0f)) + r.x;
    v.y = ((v.y > 0.f) ? v.y : (__expf(v.y) - 1.0f)) + r.y;
    v.z = ((v.z > 0.f) ? v.z : (__expf(v.z) - 1.0f)) + r.z;
    v.w = ((v.w > 0.f) ? v.w : (__expf(v.w) - 1.0f)) + r.w;
    ((float4*)out)[t] = v;
}

// ---------------- graph readout: segment sums + counts (float4) ------------
__global__ void k_graph_sum(const float* __restrict__ h, const int* __restrict__ ptr,
                            float* __restrict__ gsum, float* __restrict__ gcnt) {
    int t = blockIdx.x * 256 + threadIdx.x;
    if (t >= GN * 32) return;
    int n = t >> 5, q = t & 31, d = q * 4;
    int b = ptr[n];
    float4 hv = ((const float4*)h)[t];
    float* g = gsum + b * GD + d;
    atomicAdd(g + 0, hv.x);
    atomicAdd(g + 1, hv.y);
    atomicAdd(g + 2, hv.z);
    atomicAdd(g + 3, hv.w);
    if (q == 0) atomicAdd(&gcnt[b], 1.0f);
}

// ---------------- mean -> relu -> MLP(128->64->1) ----------------
__global__ void k_mlp(const float* __restrict__ gsum, const float* __restrict__ gcnt,
                      const float* __restrict__ w0, const float* __restrict__ b0,
                      const float* __restrict__ w1, const float* __restrict__ b1,
                      float* __restrict__ y) {
    __shared__ float g[GD];
    __shared__ float hid[GD / 2];
    int b = blockIdx.x;
    float cnt = fmaxf(gcnt[b], 1.0f);
    for (int d = threadIdx.x; d < GD; d += 64)
        g[d] = fmaxf(gsum[b * GD + d] / cnt, 0.0f);
    __syncthreads();
    int j = threadIdx.x;                       // 64 threads
    float s = b0[j];
    for (int d = 0; d < GD; ++d) s += g[d] * w0[d * (GD / 2) + j];
    hid[j] = fmaxf(s, 0.0f);
    __syncthreads();
    if (j == 0) {
        float yy = b1[0];
        for (int k = 0; k < GD / 2; ++k) yy += hid[k] * w1[k];
        y[b] = yy;
    }
}

extern "C" void kernel_launch(void* const* d_in, const int* in_sizes, int n_in,
                              void* d_out, int out_size, void* d_ws, size_t ws_size,
                              hipStream_t stream) {
    (void)in_sizes; (void)n_in; (void)out_size; (void)ws_size;
    const int*   x    = (const int*)d_in[0];
    const int*   ei   = (const int*)d_in[1];   // [2,E] row-major: src row 0, dst row 1
    const int*   ptr  = (const int*)d_in[2];
    const float* emb  = (const float*)d_in[3];
    const float* linw = (const float*)d_in[4]; // [3,128,128]
    const float* asrc = (const float*)d_in[5]; // [3,4,32]
    const float* adst = (const float*)d_in[6];
    const float* w0   = (const float*)d_in[7];
    const float* b0   = (const float*)d_in[8];
    const float* w1   = (const float*)d_in[9];
    const float* b1   = (const float*)d_in[10];
    float* y = (float*)d_out;

    // ---- carve workspace (≈ 91 MB) ----
    char* w = (char*)d_ws;
    auto carve = [&](size_t bytes) { char* p = w; w += (bytes + 255) & ~(size_t)255; return p; };
    float*     hA   = (float*)carve((size_t)GN * GD * 4);
    float*     hB   = (float*)carve((size_t)GN * GD * 4);   // scatter accum / next h
    float*     z    = (float*)carve((size_t)GN * GD * 4);
    float*     a_s  = (float*)carve((size_t)GN * GH * 4);
    float*     a_d  = (float*)carve((size_t)GN * GH * 4);
    unsigned*  amax = (unsigned*)carve((size_t)GN * GH * 4);
    float*     den  = (float*)carve((size_t)GN * GH * 4);
    float*     exv  = (float*)carve((size_t)GEP * GH * 4);
    float*     gsum = (float*)carve((size_t)GB * GD * 4);
    float*     gcnt = (float*)carve((size_t)GB * 4);
    _Float16*  wt16 = (_Float16*)carve((size_t)GD * GD * 2); // transposed f16 weight

    const int N4_B  = (GN * 32 + 255) / 256;   // float4 kernels: 6250 blocks
    const int NH_B  = (GN * GH + 255) / 256;
    const int EH_B  = (GEP * GH + 255) / 256;  // 10157 blocks
    const int E4_B  = (GEP * 32 + 255) / 256;  // 81250 blocks
    const int WD_B  = (GD * GD + 255) / 256;   // 64 blocks

    k_init_h<<<N4_B, 256, 0, stream>>>(x, emb, hA);

    float* hc = hA;   // current h
    float* ho = hB;   // accumulation target / next h
    for (int l = 0; l < 3; ++l) {
        const float* Wl = linw + (size_t)l * GD * GD;
        const float* al = asrc + (size_t)l * GH * GDH;
        const float* dl = adst + (size_t)l * GH * GDH;

        hipMemsetAsync(ho,   0, (size_t)GN * GD * 4, stream);
        hipMemsetAsync(amax, 0, (size_t)GN * GH * 4, stream); // enc-space minimum
        hipMemsetAsync(den,  0, (size_t)GN * GH * 4, stream);

        k_cvt_wt<<<WD_B, 256, 0, stream>>>(Wl, wt16);
        k_gemm_z<<<GN / 16, 256, 0, stream>>>(hc, wt16, z);
        k_attn_logits<<<NH_B, 256, 0, stream>>>(z, al, dl, a_s, a_d);
        k_edge_max<<<EH_B, 256, 0, stream>>>(ei, a_s, a_d, amax);
        k_edge_exp<<<EH_B, 256, 0, stream>>>(ei, a_s, a_d, amax, den, exv);
        k_edge_coef<<<EH_B, 256, 0, stream>>>(ei, den, exv);
        k_edge_scatter<<<E4_B, 256, 0, stream>>>(ei, z, exv, ho);
        k_elu_res<<<N4_B, 256, 0, stream>>>(ho, hc);

        float* tmp = hc; hc = ho; ho = tmp;    // ping-pong
    }

    hipMemsetAsync(gsum, 0, (size_t)GB * GD * 4, stream);
    hipMemsetAsync(gcnt, 0, (size_t)GB * 4, stream);
    k_graph_sum<<<N4_B, 256, 0, stream>>>(hc, ptr, gsum, gcnt);
    k_mlp<<<GB, 64, 0, stream>>>(gsum, gcnt, w0, b0, w1, b1, y);
}
